// SequenceLayer_46231027974769
// MI455X (gfx1250) — matrix operand order repaired and checked by reading, not compiled, not run on hardware
//
#include <hip/hip_runtime.h>
#include <hip/hip_bf16.h>
#include <math.h>

// Problem constants (match reference)
#define B_SZ   16
#define SEQ_L  4096
#define DIM_H  256
#define DIM_P  256
#define BLROWS (B_SZ * SEQ_L)   // 65536 total sequence rows

typedef __bf16 bf16;
typedef __attribute__((ext_vector_type(16))) __bf16 v16bf;
typedef __attribute__((ext_vector_type(8)))  __bf16 v8bf;
typedef __attribute__((ext_vector_type(8)))  float  v8f;
typedef __attribute__((ext_vector_type(4)))  float  v4f;

// ---------------- workspace layout (bytes) ----------------
static constexpr size_t OFF_WREP  = 0;                                    // Bc_re packed   [8][16][32][16] bf16
static constexpr size_t OFF_WIMP  = OFF_WREP  + (size_t)DIM_H*DIM_P*2;    // Bc_im packed
static constexpr size_t OFF_WCREP = OFF_WIMP  + (size_t)DIM_H*DIM_P*2;    // C_re  packed
static constexpr size_t OFF_WCIMP = OFF_WCREP + (size_t)DIM_P*DIM_H*2;    // -C_im packed
static constexpr size_t OFF_ARE   = OFF_WCIMP + (size_t)DIM_P*DIM_H*2;    // Lam_bar re [P] f32
static constexpr size_t OFF_AIM   = OFF_ARE   + (size_t)DIM_P*4;          // Lam_bar im [P] f32
static constexpr size_t OFF_BU    = (size_t)1 << 20;                      // Bu packed 2xbf16 (u32) [BL][P]
static constexpr size_t OFF_SRE   = OFF_BU  + (size_t)BLROWS*DIM_P*4;     // states re bf16 [BL][P]
static constexpr size_t OFF_SIM   = OFF_SRE + (size_t)BLROWS*DIM_P*2;     // states im bf16 [BL][P]

__device__ __forceinline__ unsigned int pack_bf16x2(float re, float im) {
    unsigned short ur = __builtin_bit_cast(unsigned short, (bf16)re);
    unsigned short ui = __builtin_bit_cast(unsigned short, (bf16)im);
    return (unsigned int)ur | ((unsigned int)ui << 16);
}

__device__ __forceinline__ v16bf load_frag(const bf16* p) {
    return *(const v16bf*)p;   // 32B contiguous per lane -> 2x global_load_b128
}

// ---------------- kernel 0: discretization + fragment-order weight repack ----------------
// Packed 32(K)x16(N) WMMA-B tile: idx = ((ktile*16 + ntile)*32 + lane)*16 + e,
// lane = kh*16 + n, element e -> K = kh*8 + (e&7) + (e>>3)*16 (ISA 16-bit B-matrix layout).
__global__ __launch_bounds__(256)
void s5_prep_kernel(const float* __restrict__ Lre, const float* __restrict__ Lim,
                    const float* __restrict__ Bre, const float* __restrict__ Bim,
                    const float* __restrict__ Cre, const float* __restrict__ Cim,
                    const float* __restrict__ lst,
                    bf16* __restrict__ WreP, bf16* __restrict__ WimP,
                    bf16* __restrict__ WCreP, bf16* __restrict__ WCimP,
                    float* __restrict__ Are, float* __restrict__ Aim)
{
    int idx = blockIdx.x * blockDim.x + threadIdx.x;   // over P*H
    int p = idx >> 8;          // DIM_H == 256
    int h = idx & 255;
    float dt    = expf(lst[p]);
    float lam_r = -fabsf(Lre[p]);       // stability 'abs'
    float lam_i = Lim[p];
    float a = lam_r * dt, b = lam_i * dt;
    float ea = expf(a);
    float er = ea * cosf(b);            // Lam_bar
    float ei = ea * sinf(b);
    // gamma = (Lam_bar - 1) / Lam
    float den = lam_r*lam_r + lam_i*lam_i;
    float nr = er - 1.0f, ni = ei;
    float gr = (nr*lam_r + ni*lam_i) / den;
    float gi = (ni*lam_r - nr*lam_i) / den;
    float br = Bre[p*DIM_H + h], bi = Bim[p*DIM_H + h];

    // Input GEMM B-operand: K = h, N = p, value = Bc[p][h]
    {
        int kt = h >> 5, kh = (h >> 3) & 1, e = (h & 7) | ((h >> 1) & 8);
        int pt = p >> 4, n  = p & 15;
        size_t o = ((size_t)(kt*16 + pt)*32 + (kh*16 + n))*16 + e;
        WreP[o] = (bf16)(br*gr - bi*gi);
        WimP[o] = (bf16)(br*gi + bi*gr);
    }
    // Output GEMM B-operand: K = p, N = h, value = C_re[h,p] / -C_im[h,p]
    {
        int kt = p >> 5, kh = (p >> 3) & 1, e = (p & 7) | ((p >> 1) & 8);
        int ht = h >> 4, n  = h & 15;
        size_t o = ((size_t)(kt*16 + ht)*32 + (kh*16 + n))*16 + e;
        WCreP[o] = (bf16)( Cre[(size_t)h*DIM_P + p]);
        WCimP[o] = (bf16)(-Cim[(size_t)h*DIM_P + p]);
    }
    if (h == 0) { Are[p] = er; Aim[p] = ei; }
}

// ---------------- kernel 1: Bu = x @ Bc^T  (bf16 WMMA, f32 accumulate) ----------------
// 256 threads = 8 waves; tile M=128 (wave owns 16 rows) x N=64; K stepped 32 over H.
// No LDS. A: two contiguous 8-float runs per lane (x is K-contiguous) + packed cvt.
// B: one 32B per-lane load from fragment-packed weights, double-buffered over nt.
// A-operand loads for the next k-step are issued before the current WMMA burst.
__global__ __launch_bounds__(256)
void s5_gemm_in_kernel(const float* __restrict__ x,
                       const bf16* __restrict__ WreP, const bf16* __restrict__ WimP,
                       unsigned int* __restrict__ BuPk)
{
    const int tid  = threadIdx.x;
    const int w    = tid >> 5;
    const int lane = tid & 31;
    const int r    = lane & 15;
    const int hi   = lane >> 4;
    const int l0   = blockIdx.x * 128;
    const int p0   = blockIdx.y * 64;
    const int row  = l0 + w*16 + r;

    v8f accRe[4], accIm[4];
    #pragma unroll
    for (int nt = 0; nt < 4; ++nt) {
        v8f z = {0.f,0.f,0.f,0.f,0.f,0.f,0.f,0.f};
        accRe[nt] = z; accIm[nt] = z;
    }

    const float* xrow = x + (size_t)row*DIM_H + hi*8;
    v4f f0 = *(const v4f*)(xrow +  0);
    v4f f1 = *(const v4f*)(xrow +  4);
    v4f f2 = *(const v4f*)(xrow + 16);
    v4f f3 = *(const v4f*)(xrow + 20);

    for (int kt8 = 0; kt8 < 8; ++kt8) {
        // A fragment (16x32): elements 0-7 -> K=hi*8+e, 8-15 -> K=16+hi*8+(e-8)
        v8bf alo, ahi;
        #pragma unroll
        for (int j = 0; j < 4; ++j) {
            alo[j]   = (bf16)f0[j];
            alo[4+j] = (bf16)f1[j];
            ahi[j]   = (bf16)f2[j];
            ahi[4+j] = (bf16)f3[j];
        }
        v16bf afrag = __builtin_shufflevector(alo, ahi,
                        0,1,2,3,4,5,6,7,8,9,10,11,12,13,14,15);

        if (kt8 < 7) {                      // issue next k-step A loads early
            const float* xn = xrow + (kt8 + 1)*32;
            f0 = *(const v4f*)(xn +  0);
            f1 = *(const v4f*)(xn +  4);
            f2 = *(const v4f*)(xn + 16);
            f3 = *(const v4f*)(xn + 20);
        }

        const size_t bo = (((size_t)kt8*16 + (p0 >> 4))*32 + lane)*16;
        v16bf bR = load_frag(WreP + bo);
        v16bf bI = load_frag(WimP + bo);
        #pragma unroll
        for (int nt = 0; nt < 4; ++nt) {
            v16bf bRn, bIn;
            if (nt < 3) {                   // double-buffer next ntile (stride 512 elems)
                bRn = load_frag(WreP + bo + (size_t)(nt + 1)*512);
                bIn = load_frag(WimP + bo + (size_t)(nt + 1)*512);
            }
            accRe[nt] = __builtin_amdgcn_wmma_f32_16x16x32_bf16(
                            false, afrag, false, bR, (short)0, accRe[nt], false, false);
            accIm[nt] = __builtin_amdgcn_wmma_f32_16x16x32_bf16(
                            false, afrag, false, bI, (short)0, accIm[nt], false, false);
            if (nt < 3) { bR = bRn; bI = bIn; }
        }
    }

    // D layout: element v -> row hi*8+v, col = lane&15; pack re/im as 2xbf16 in u32.
    #pragma unroll
    for (int nt = 0; nt < 4; ++nt) {
        #pragma unroll
        for (int v = 0; v < 8; ++v) {
            size_t gl = (size_t)(l0 + w*16 + hi*8 + v);
            int    gp = p0 + nt*16 + r;
            BuPk[gl*DIM_P + gp] = pack_bf16x2(accRe[nt][v], accIm[nt][v]);
        }
    }
}

// ---------------- kernel 2: diagonal complex recurrence (constant A per channel) ----------
// One thread per (b,p) chain; a wave covers 32 consecutive p -> one 128B line per step.
__global__ __launch_bounds__(256)
void s5_scan_kernel(const unsigned int* __restrict__ Bu,
                    const float* __restrict__ Are, const float* __restrict__ Aim,
                    bf16* __restrict__ sre, bf16* __restrict__ sim)
{
    int tid = blockIdx.x * blockDim.x + threadIdx.x;   // 0 .. B*P-1
    int b = tid >> 8;
    int p = tid & 255;
    const float ar = Are[p], ai = Aim[p];
    float s_r = 0.f, s_i = 0.f;
    size_t base = ((size_t)b * SEQ_L) * DIM_P + p;
    for (int l = 0; l < SEQ_L; l += 4) {
        #pragma unroll
        for (int j = 0; j < 4; ++j) {
            size_t o = base + (size_t)(l + j) * DIM_P;
            __builtin_prefetch(&Bu[o + 32*DIM_P], 0, 3);   // global_prefetch_b8, 32 steps ahead
            unsigned int u = Bu[o];
            float ur = (float)__builtin_bit_cast(bf16, (unsigned short)(u & 0xffffu));
            float ui = (float)__builtin_bit_cast(bf16, (unsigned short)(u >> 16));
            float nrl = fmaf(ar, s_r, fmaf(-ai, s_i, ur));
            float nil = fmaf(ar, s_i, fmaf( ai, s_r, ui));
            s_r = nrl; s_i = nil;
            sre[o] = (bf16)nrl;
            sim[o] = (bf16)nil;
        }
    }
}

// ---------------- kernel 3: y = s_re@C_re^T + s_im@(-C_im^T); out = relu(y)+x ------------
__global__ __launch_bounds__(256)
void s5_gemm_out_kernel(const bf16* __restrict__ sre, const bf16* __restrict__ sim,
                        const bf16* __restrict__ WCreP, const bf16* __restrict__ WCimP,
                        const float* __restrict__ x, float* __restrict__ out)
{
    const int tid  = threadIdx.x;
    const int w    = tid >> 5;
    const int lane = tid & 31;
    const int r    = lane & 15;
    const int hi   = lane >> 4;
    const int l0   = blockIdx.x * 128;
    const int h0   = blockIdx.y * 64;
    const int row  = l0 + w*16 + r;

    v8f acc[4];
    #pragma unroll
    for (int nt = 0; nt < 4; ++nt) {
        v8f z = {0.f,0.f,0.f,0.f,0.f,0.f,0.f,0.f};
        acc[nt] = z;
    }

    const bf16* srow = sre + (size_t)row*DIM_P + hi*8;
    const bf16* irow = sim + (size_t)row*DIM_P + hi*8;
    v8bf r0 = *(const v8bf*)(srow);
    v8bf r1 = *(const v8bf*)(srow + 16);
    v8bf i0 = *(const v8bf*)(irow);
    v8bf i1 = *(const v8bf*)(irow + 16);

    for (int kt8 = 0; kt8 < 8; ++kt8) {
        v16bf aR = __builtin_shufflevector(r0, r1, 0,1,2,3,4,5,6,7,8,9,10,11,12,13,14,15);
        v16bf aI = __builtin_shufflevector(i0, i1, 0,1,2,3,4,5,6,7,8,9,10,11,12,13,14,15);

        if (kt8 < 7) {                      // issue next k-step A loads early
            const bf16* sn = srow + (kt8 + 1)*32;
            const bf16* in_ = irow + (kt8 + 1)*32;
            r0 = *(const v8bf*)(sn);
            r1 = *(const v8bf*)(sn + 16);
            i0 = *(const v8bf*)(in_);
            i1 = *(const v8bf*)(in_ + 16);
        }

        const size_t bo = (((size_t)kt8*16 + (h0 >> 4))*32 + lane)*16;
        v16bf bR = load_frag(WCreP + bo);
        v16bf bI = load_frag(WCimP + bo);
        #pragma unroll
        for (int nt = 0; nt < 4; ++nt) {
            v16bf bRn, bIn;
            if (nt < 3) {
                bRn = load_frag(WCreP + bo + (size_t)(nt + 1)*512);
                bIn = load_frag(WCimP + bo + (size_t)(nt + 1)*512);
            }
            acc[nt] = __builtin_amdgcn_wmma_f32_16x16x32_bf16(
                          false, aR, false, bR, (short)0, acc[nt], false, false);
            acc[nt] = __builtin_amdgcn_wmma_f32_16x16x32_bf16(
                          false, aI, false, bI, (short)0, acc[nt], false, false);
            if (nt < 3) { bR = bRn; bI = bIn; }
        }
    }

    #pragma unroll
    for (int nt = 0; nt < 4; ++nt) {
        #pragma unroll
        for (int v = 0; v < 8; ++v) {
            size_t gl = (size_t)(l0 + w*16 + hi*8 + v);
            int    gh = h0 + nt*16 + r;
            float y = acc[nt][v];
            out[gl*DIM_H + gh] = fmaxf(y, 0.0f) + x[gl*DIM_H + gh];   // relu + residual
        }
    }
}

// ---------------- host launcher ----------------
extern "C" void kernel_launch(void* const* d_in, const int* in_sizes, int n_in,
                              void* d_out, int out_size, void* d_ws, size_t ws_size,
                              hipStream_t stream)
{
    (void)in_sizes; (void)n_in; (void)out_size; (void)ws_size;
    const float* x   = (const float*)d_in[0];
    const float* Lre = (const float*)d_in[1];
    const float* Lim = (const float*)d_in[2];
    const float* Bre = (const float*)d_in[3];
    const float* Bim = (const float*)d_in[4];
    const float* Cre = (const float*)d_in[5];
    const float* Cim = (const float*)d_in[6];
    const float* lst = (const float*)d_in[7];

    char* ws = (char*)d_ws;
    bf16*         WreP  = (bf16*)        (ws + OFF_WREP);
    bf16*         WimP  = (bf16*)        (ws + OFF_WIMP);
    bf16*         WCreP = (bf16*)        (ws + OFF_WCREP);
    bf16*         WCimP = (bf16*)        (ws + OFF_WCIMP);
    float*        Are   = (float*)       (ws + OFF_ARE);
    float*        Aim   = (float*)       (ws + OFF_AIM);
    unsigned int* BuPk  = (unsigned int*)(ws + OFF_BU);
    bf16*         sre   = (bf16*)        (ws + OFF_SRE);
    bf16*         sim   = (bf16*)        (ws + OFF_SIM);

    s5_prep_kernel<<<(DIM_P*DIM_H)/256, 256, 0, stream>>>(
        Lre, Lim, Bre, Bim, Cre, Cim, lst, WreP, WimP, WCreP, WCimP, Are, Aim);

    s5_gemm_in_kernel<<<dim3(BLROWS/128, DIM_P/64), 256, 0, stream>>>(x, WreP, WimP, BuPk);

    s5_scan_kernel<<<(B_SZ*DIM_P)/256, 256, 0, stream>>>(BuPk, Are, Aim, sre, sim);

    s5_gemm_out_kernel<<<dim3(BLROWS/128, DIM_H/64), 256, 0, stream>>>(
        sre, sim, WCreP, WCimP, x, (float*)d_out);
}
